// T5LayerSelfAttention_4690104287575
// MI455X (gfx1250) — matrix operand order repaired and checked by reading, not compiled, not run on hardware
//
#include <hip/hip_runtime.h>
#include <math.h>

#define D_MODEL 512
#define N_HEADS 8
#define D_KV 64
#define SEQ 2048
#define NBATCH 8
#define NS (NBATCH * SEQ) // 16384

typedef __attribute__((ext_vector_type(16))) __bf16 v16bf;
typedef __attribute__((ext_vector_type(8)))  __bf16 v8bf;
typedef __attribute__((ext_vector_type(8)))  float  v8f;

union AF { v16bf v; v8bf h[2]; };

static __device__ __forceinline__ __bf16 f2bf(float f) {
  union { float f; unsigned u; } a; a.f = f;
  unsigned r = a.u + 0x7FFFu + ((a.u >> 16) & 1u);
  union { unsigned short s; __bf16 b; } o; o.s = (unsigned short)(r >> 16);
  return o.b;
}

static __device__ __forceinline__ v8f wmma_bf16(v16bf a, v16bf b, v8f c) {
  // D = A(16x32 bf16) * B(32x16 bf16) + C(16x16 f32)
  return __builtin_amdgcn_wmma_f32_16x16x32_bf16(false, a, false, b, (short)0, c, false, false);
}

// A fragment (16x32) from row-major bf16 [*, ld] at (row0, k0).
// Lane layout: row = lane%16; e<8 -> k = k0+8*half+e ; e>=8 -> k = k0+16+8*half+(e-8)
static __device__ __forceinline__ v16bf load_afrag(const __bf16* __restrict__ X, int ld,
                                                   int row0, int k0, int lr, int half) {
  const __bf16* p = X + (size_t)(row0 + lr) * ld + k0 + 8 * half;
  AF a;
  a.h[0] = *(const v8bf*)p;
  a.h[1] = *(const v8bf*)(p + 16);
  return a.v;
}

// B fragment: logical B[k][n] (32x16) at (k0, n0), storage row-major [n][k] stride ld.
// Lane layout: col = lane%16; e<8 -> k = k0+8*half+e ; e>=8 -> k = k0+16+8*half+(e-8)
static __device__ __forceinline__ v16bf load_bfrag(const __bf16* __restrict__ Bt, int ld,
                                                   int k0, int n0, int lr, int half) {
  const __bf16* p = Bt + (size_t)(n0 + lr) * ld + k0 + 8 * half;
  AF b;
  b.h[0] = *(const v8bf*)p;
  b.h[1] = *(const v8bf*)(p + 16);
  return b.v;
}

// T5 bidirectional relative-position bucket (num_buckets=32, max_distance=128)
static __device__ __forceinline__ int rel_bucket(int rel) {
  int bucket = (rel > 0) ? 16 : 0;
  int ar = rel < 0 ? -rel : rel;
  if (ar < 8) return bucket + ar;
  float relf = (float)ar;
  int large = 8 + (int)(__logf(relf * 0.125f) * (8.0f / __logf(16.0f)));
  if (large > 15) large = 15;
  return bucket + large;
}

// ---------------- RMS norm -> bf16 ----------------
__global__ void __launch_bounds__(256)
k_rmsnorm(const float* __restrict__ x, const float* __restrict__ w, __bf16* __restrict__ out) {
  __shared__ float red[256];
  int row = blockIdx.x;
  int t = threadIdx.x;
  const float* xr = x + (size_t)row * D_MODEL;
  float a0 = xr[t], a1 = xr[t + 256];
  red[t] = a0 * a0 + a1 * a1;
  __syncthreads();
  for (int s = 128; s > 0; s >>= 1) {
    if (t < s) red[t] += red[t + s];
    __syncthreads();
  }
  float rs = rsqrtf(red[0] * (1.0f / D_MODEL) + 1e-6f);
  __bf16* o = out + (size_t)row * D_MODEL;
  o[t]       = f2bf(a0 * rs * w[t]);
  o[t + 256] = f2bf(a1 * rs * w[t + 256]);
}

// ---------------- weights: f32 [k][n] -> bf16 transposed [n][k], 4 matrices ----------------
__global__ void __launch_bounds__(256)
k_wconv(const float* __restrict__ wq, const float* __restrict__ wk,
        const float* __restrict__ wv, const float* __restrict__ wo,
        __bf16* __restrict__ out) {
  int idx = blockIdx.x * 256 + threadIdx.x;     // < 4 * 2^18
  int wsel = idx >> 18;
  int rem = idx & 262143;
  int k = rem >> 9;
  int n = rem & 511;
  const float* src = (wsel == 0) ? wq : (wsel == 1) ? wk : (wsel == 2) ? wv : wo;
  out[((size_t)wsel << 18) + ((size_t)n << 9) + k] = f2bf(src[rem]);
}

// ---------------- QKV GEMM: Y = normed(bf16) @ W(bf16), WMMA ----------------
// z=0 -> Q[N,H,S,D]; z=1 -> K[N,H,S,D]; z=2 -> V^T [N,H,D,S]
__global__ void __launch_bounds__(256)
k_qkv(const __bf16* __restrict__ Xn, const __bf16* __restrict__ Wt,
      __bf16* __restrict__ Qo, __bf16* __restrict__ Ko, __bf16* __restrict__ Vt) {
  int lane = threadIdx.x & 31, wv_ = threadIdx.x >> 5;
  int lr = lane & 15, half = lane >> 4;
  int z = blockIdx.z;
  int m0 = blockIdx.y * 128 + wv_ * 16;
  int n0 = blockIdx.x * 64;
  const __bf16* W = Wt + ((size_t)z << 18);
  v8f acc[4] = {{}, {}, {}, {}};
  for (int k0 = 0; k0 < 512; k0 += 32) {
    v16bf a = load_afrag(Xn, 512, m0, k0, lr, half);
    acc[0] = wmma_bf16(a, load_bfrag(W, 512, k0, n0 + 0,  lr, half), acc[0]);
    acc[1] = wmma_bf16(a, load_bfrag(W, 512, k0, n0 + 16, lr, half), acc[1]);
    acc[2] = wmma_bf16(a, load_bfrag(W, 512, k0, n0 + 32, lr, half), acc[2]);
    acc[3] = wmma_bf16(a, load_bfrag(W, 512, k0, n0 + 48, lr, half), acc[3]);
  }
#pragma unroll
  for (int b = 0; b < 4; ++b) {
    int col = n0 + b * 16 + lr;
    int h = col >> 6, d = col & 63;
#pragma unroll
    for (int r = 0; r < 8; ++r) {
      int m = m0 + r + 8 * half;
      int nb = m >> 11;
      int s = m & 2047;
      __bf16 val = f2bf(acc[b][r]);
      if (z == 0)      Qo[((((size_t)nb * 8 + h) * 2048 + s) << 6) + d] = val;
      else if (z == 1) Ko[((((size_t)nb * 8 + h) * 2048 + s) << 6) + d] = val;
      else             Vt[(((size_t)nb * 8 + h) * 64 + d) * 2048 + s] = val;
    }
  }
}

// ---------------- flash attention, 1 wave per 16 query rows ----------------
__global__ void __launch_bounds__(128)
k_attn(const __bf16* __restrict__ Q, const __bf16* __restrict__ K,
       const __bf16* __restrict__ Vt, const float* __restrict__ table,
       __bf16* __restrict__ ctx) {
  __shared__ __align__(16) __bf16 Pbuf[4][16 * 32];
  __shared__ float btab[32];
  int lane = threadIdx.x & 31, w = threadIdx.x >> 5;
  int lr = lane & 15, half = lane >> 4;
  int nh = blockIdx.y;          // nb*8 + h
  int h = nh & 7;
  int q0 = blockIdx.x * 64 + w * 16;
  if (threadIdx.x < 32) btab[threadIdx.x] = table[threadIdx.x * N_HEADS + h];
  __syncthreads();

  const __bf16* Qb = Q + ((size_t)nh * 2048) * 64;
  const __bf16* Kb = K + ((size_t)nh * 2048) * 64;
  const __bf16* Vb = Vt + ((size_t)nh * 64) * 2048;

  v16bf aq0 = load_afrag(Qb, 64, q0, 0, lr, half);
  v16bf aq1 = load_afrag(Qb, 64, q0, 32, lr, half);

  float mrow[8], lrow[8];
#pragma unroll
  for (int r = 0; r < 8; ++r) { mrow[r] = -1e30f; lrow[r] = 0.0f; }
  v8f acc0 = {}, acc1 = {}, acc2 = {}, acc3 = {};
  __bf16* P = &Pbuf[w][0];

  for (int kb = 0; kb < 2048; kb += 32) {
    // scores: tile0 = keys kb..kb+15, tile1 = kb+16..kb+31 ; contraction over d (K=64)
    v8f c0 = {}, c1 = {};
    c0 = wmma_bf16(aq0, load_bfrag(Kb, 64, 0,  kb,      lr, half), c0);
    c0 = wmma_bf16(aq1, load_bfrag(Kb, 64, 32, kb,      lr, half), c0);
    c1 = wmma_bf16(aq0, load_bfrag(Kb, 64, 0,  kb + 16, lr, half), c1);
    c1 = wmma_bf16(aq1, load_bfrag(Kb, 64, 32, kb + 16, lr, half), c1);
    int key0 = kb + lr, key1 = kb + 16 + lr;
#pragma unroll
    for (int r = 0; r < 8; ++r) {
      int q = q0 + r + 8 * half;
      float s0 = c0[r] + btab[rel_bucket(key0 - q)];
      float s1 = c1[r] + btab[rel_bucket(key1 - q)];
      // row max over the 16 lanes of this half (xor<16 preserves half bit)
      float tm = fmaxf(s0, s1);
      tm = fmaxf(tm, __shfl_xor(tm, 1, 32));
      tm = fmaxf(tm, __shfl_xor(tm, 2, 32));
      tm = fmaxf(tm, __shfl_xor(tm, 4, 32));
      tm = fmaxf(tm, __shfl_xor(tm, 8, 32));
      float mnew = fmaxf(mrow[r], tm);
      float scale = __expf(mrow[r] - mnew);
      float p0 = __expf(s0 - mnew);
      float p1 = __expf(s1 - mnew);
      float ps = p0 + p1;
      ps += __shfl_xor(ps, 1, 32);
      ps += __shfl_xor(ps, 2, 32);
      ps += __shfl_xor(ps, 4, 32);
      ps += __shfl_xor(ps, 8, 32);
      lrow[r] = lrow[r] * scale + ps;
      mrow[r] = mnew;
      acc0[r] *= scale; acc1[r] *= scale; acc2[r] *= scale; acc3[r] *= scale;
      // stash P tile (C layout -> LDS) for re-read in A layout; same wave, DS in-order
      int prow = (r + 8 * half) * 32;
      P[prow + lr]      = f2bf(p0);
      P[prow + 16 + lr] = f2bf(p1);
    }
    // P(16x32) @ V(32x64): A-frag from LDS, B-frags straight from V^T
    AF pa;
    pa.h[0] = *(const v8bf*)(P + lr * 32 + 8 * half);
    pa.h[1] = *(const v8bf*)(P + lr * 32 + 16 + 8 * half);
    acc0 = wmma_bf16(pa.v, load_bfrag(Vb, 2048, kb, 0,  lr, half), acc0);
    acc1 = wmma_bf16(pa.v, load_bfrag(Vb, 2048, kb, 16, lr, half), acc1);
    acc2 = wmma_bf16(pa.v, load_bfrag(Vb, 2048, kb, 32, lr, half), acc2);
    acc3 = wmma_bf16(pa.v, load_bfrag(Vb, 2048, kb, 48, lr, half), acc3);
  }
  int nb = nh >> 3;
#pragma unroll
  for (int r = 0; r < 8; ++r) {
    float inv = 1.0f / lrow[r];
    int q = q0 + r + 8 * half;
    __bf16* o = ctx + ((size_t)(nb * 2048 + q)) * 512 + h * 64 + lr;
    o[0]  = f2bf(acc0[r] * inv);
    o[16] = f2bf(acc1[r] * inv);
    o[32] = f2bf(acc2[r] * inv);
    o[48] = f2bf(acc3[r] * inv);
  }
}

// ---------------- output projection + residual ----------------
__global__ void __launch_bounds__(256)
k_out(const __bf16* __restrict__ Ctx, const __bf16* __restrict__ Wot,
      const float* __restrict__ hid, float* __restrict__ out) {
  int lane = threadIdx.x & 31, w = threadIdx.x >> 5;
  int lr = lane & 15, half = lane >> 4;
  int m0 = blockIdx.y * 128 + w * 16;
  int n0 = blockIdx.x * 64;
  v8f acc[4] = {{}, {}, {}, {}};
  for (int k0 = 0; k0 < 512; k0 += 32) {
    v16bf a = load_afrag(Ctx, 512, m0, k0, lr, half);
    acc[0] = wmma_bf16(a, load_bfrag(Wot, 512, k0, n0 + 0,  lr, half), acc[0]);
    acc[1] = wmma_bf16(a, load_bfrag(Wot, 512, k0, n0 + 16, lr, half), acc[1]);
    acc[2] = wmma_bf16(a, load_bfrag(Wot, 512, k0, n0 + 32, lr, half), acc[2]);
    acc[3] = wmma_bf16(a, load_bfrag(Wot, 512, k0, n0 + 48, lr, half), acc[3]);
  }
#pragma unroll
  for (int b = 0; b < 4; ++b) {
    int col = n0 + b * 16 + lr;
#pragma unroll
    for (int r = 0; r < 8; ++r) {
      size_t idx = (size_t)(m0 + r + 8 * half) * 512 + col;
      out[idx] = hid[idx] + acc[b][r];
    }
  }
}

// ---------------- position bias [1,H,S,S] + remain_node_idx ----------------
__global__ void __launch_bounds__(256)
k_bias(const float* __restrict__ table, float* __restrict__ outb, float* __restrict__ outr) {
  unsigned idx = blockIdx.x * 256u + threadIdx.x; // < 8*2048*2048
  int h = (int)(idx >> 22);
  int rem = (int)(idx & 4194303u);
  int q = rem >> 11;
  int k = rem & 2047;
  outb[idx] = table[rel_bucket(k - q) * N_HEADS + h];
  if (idx < 8) outr[idx] = (float)idx;
}

extern "C" void kernel_launch(void* const* d_in, const int* in_sizes, int n_in,
                              void* d_out, int out_size, void* d_ws, size_t ws_size,
                              hipStream_t stream) {
  (void)in_sizes; (void)n_in; (void)out_size; (void)ws_size;
  const float* hidden = (const float*)d_in[0];
  const float* lnw    = (const float*)d_in[1];
  const float* wq     = (const float*)d_in[2];
  const float* wk     = (const float*)d_in[3];
  const float* wv     = (const float*)d_in[4];
  const float* wo     = (const float*)d_in[5];
  const float* table  = (const float*)d_in[6];
  // d_in[7] seq_idx2graph_idx: only its length (8) matters

  char* ws = (char*)d_ws;
  __bf16* normed = (__bf16*)ws;                              // 16 MB
  __bf16* Wt     = (__bf16*)(ws + (size_t)(16u << 20));      // 2 MB (q,k,v,o transposed bf16)
  __bf16* Qo     = (__bf16*)(ws + (size_t)(18u << 20));      // 16 MB [N,H,S,D]
  __bf16* Ko     = (__bf16*)(ws + (size_t)(34u << 20));      // 16 MB [N,H,S,D]
  __bf16* Vt     = (__bf16*)(ws + (size_t)(50u << 20));      // 16 MB [N,H,D,S]
  __bf16* Ctx    = normed;                                   // reuse: normed dead after QKV

  float* out_hidden = (float*)d_out;                         // [8,2048,512]
  float* out_bias   = out_hidden + (size_t)NS * D_MODEL;     // [1,8,2048,2048]
  float* out_remain = out_bias + (size_t)N_HEADS * SEQ * SEQ;

  k_rmsnorm<<<NS, 256, 0, stream>>>(hidden, lnw, normed);
  k_wconv<<<4096, 256, 0, stream>>>(wq, wk, wv, wo, Wt);
  dim3 g1(8, 128, 3);
  k_qkv<<<g1, 256, 0, stream>>>(normed, Wt, Qo, Ko, Vt);
  dim3 g2(32, 64);
  k_attn<<<g2, 128, 0, stream>>>(Qo, Ko, Vt, table, Ctx);
  dim3 g3(8, 128);
  k_out<<<g3, 256, 0, stream>>>(Ctx, Wt + 3 * (512 * 512), hidden, out_hidden);
  k_bias<<<131072, 256, 0, stream>>>(table, out_bias, out_remain);
}
